// MultiHeadAttention_60859686584815
// MI455X (gfx1250) — matrix operand order repaired and checked by reading, not compiled, not run on hardware
//
#include <hip/hip_runtime.h>

// ---------------------------------------------------------------------------
// MI455X / gfx1250 multi-head-attention-without-head-split, bf16 WMMA path.
// B=4, S=2048, D=1024.  All GEMMs via v_wmma_f32_16x16x32_bf16.
// ---------------------------------------------------------------------------

#define Bn 4
#define Sn 2048
#define Dn 1024

typedef __bf16 bf16_t;
typedef __attribute__((ext_vector_type(16))) __bf16 v16bf;
typedef __attribute__((ext_vector_type(8)))  __bf16 bf16x8;
typedef __attribute__((ext_vector_type(8)))  float  v8f;

// --------------------------- fp32 -> bf16 convert ---------------------------
__global__ void cvt_f32_bf16(const float* __restrict__ in,
                             bf16_t* __restrict__ out, int n) {
  int i = blockIdx.x * 256 + threadIdx.x;
  if (i < n) out[i] = (bf16_t)in[i];
}

// ------------------------------ WMMA GEMM ----------------------------------
// C[M,N] = scale * A[M,K] x B^T   where Bsrc row n supplies column n of the
// B-matrix (i.e. this computes A @ Bsrc^T with Bsrc rows contiguous in K).
//
// Per-lane fragment addressing follows CDNA5 ISA 7.12.2 (wave32):
//   A 16x32 bf16 : lane L -> row M=L%16, K chunks {h*8..h*8+7, 16+h*8..}
//   B 32x16 bf16 : lane L -> col N=L%16, K chunks {h*8..h*8+7, 16+h*8..}
//   C 16x16 f32  : VGPR r, lane L -> M = r + h*8, N = L%16      (h = L/16)
//
// MODE 0: bf16 out, row-major      O[row*ldc + col]
// MODE 1: bf16 out, transposed     O[b*outBStride + col*ldc + qlocal]
// MODE 2: f32  out, row-major, scaled
template <int MODE>
__global__ __launch_bounds__(128)
void gemm_bf16_wmma(const bf16_t* __restrict__ A, int lda,
                    const bf16_t* __restrict__ Bsrc, int ldb,
                    void* __restrict__ Out, int ldc,
                    int Kdim, float scale, int batchS, long long outBStride) {
  const int tid  = threadIdx.x;
  const int wave = tid >> 5;
  const int lane = tid & 31;
  const int half = lane >> 4;   // 0 or 1
  const int l16  = lane & 15;

  const int rowbase = blockIdx.y * 128 + (wave >> 1) * 64;  // wave: 64x64 tile
  const int colbase = blockIdx.x * 128 + (wave & 1) * 64;

  v8f acc[4][4];
  {
    v8f z = {};
#pragma unroll
    for (int i = 0; i < 4; ++i)
#pragma unroll
      for (int j = 0; j < 4; ++j) acc[i][j] = z;
  }

  for (int k0 = 0; k0 < Kdim; k0 += 32) {
    v16bf af[4], bf[4];
#pragma unroll
    for (int mi = 0; mi < 4; ++mi) {
      const bf16_t* p =
          A + (size_t)(rowbase + mi * 16 + l16) * lda + k0 + half * 8;
      bf16x8 lo = *(const bf16x8*)p;          // K = base .. base+7
      bf16x8 hi = *(const bf16x8*)(p + 16);   // K = base+16 .. base+23
      af[mi] = __builtin_shufflevector(lo, hi, 0, 1, 2, 3, 4, 5, 6, 7, 8, 9,
                                       10, 11, 12, 13, 14, 15);
    }
#pragma unroll
    for (int ni = 0; ni < 4; ++ni) {
      const bf16_t* p =
          Bsrc + (size_t)(colbase + ni * 16 + l16) * ldb + k0 + half * 8;
      bf16x8 lo = *(const bf16x8*)p;
      bf16x8 hi = *(const bf16x8*)(p + 16);
      bf[ni] = __builtin_shufflevector(lo, hi, 0, 1, 2, 3, 4, 5, 6, 7, 8, 9,
                                       10, 11, 12, 13, 14, 15);
    }
#pragma unroll
    for (int mi = 0; mi < 4; ++mi)
#pragma unroll
      for (int ni = 0; ni < 4; ++ni)
        acc[mi][ni] = __builtin_amdgcn_wmma_f32_16x16x32_bf16(
            false, af[mi], false, bf[ni], (short)0, acc[mi][ni], false, false);
  }

  if (MODE == 0) {  // bf16 row-major
    bf16_t* O = (bf16_t*)Out;
#pragma unroll
    for (int mi = 0; mi < 4; ++mi) {
      const int row0 = rowbase + mi * 16 + half * 8;
#pragma unroll
      for (int ni = 0; ni < 4; ++ni) {
        const int col = colbase + ni * 16 + l16;
#pragma unroll
        for (int r = 0; r < 8; ++r)
          O[(size_t)(row0 + r) * ldc + col] = (bf16_t)(acc[mi][ni][r] * scale);
      }
    }
  } else if (MODE == 1) {  // bf16 transposed (column-major), batched
    bf16_t* O = (bf16_t*)Out;
#pragma unroll
    for (int mi = 0; mi < 4; ++mi) {
      const int qg = rowbase + mi * 16 + half * 8;
      const int bb = qg / batchS;
      const int ql = qg - bb * batchS;
#pragma unroll
      for (int ni = 0; ni < 4; ++ni) {
        const int col = colbase + ni * 16 + l16;
        bf16x8 v;
#pragma unroll
        for (int r = 0; r < 8; ++r) v[r] = (bf16_t)(acc[mi][ni][r] * scale);
        *(bf16x8*)(O + (long long)bb * outBStride + (size_t)col * ldc + ql) = v;
      }
    }
  } else {  // MODE 2: f32 row-major, scaled
    float* O = (float*)Out;
#pragma unroll
    for (int mi = 0; mi < 4; ++mi) {
      const int row0 = rowbase + mi * 16 + half * 8;
#pragma unroll
      for (int ni = 0; ni < 4; ++ni) {
        const int col = colbase + ni * 16 + l16;
#pragma unroll
        for (int r = 0; r < 8; ++r)
          O[(size_t)(row0 + r) * ldc + col] = acc[mi][ni][r] * scale;
      }
    }
  }
}

// ------------------------------ row softmax --------------------------------
// fp32 in (modified in-place with exp), bf16 out.  One 256-thread block/row.
__global__ __launch_bounds__(256)
void softmax_row(float* __restrict__ sc, bf16_t* __restrict__ attn, int N) {
  const int row = blockIdx.x;
  float* p  = sc + (size_t)row * N;
  bf16_t* o = attn + (size_t)row * N;
  __shared__ float red[256];
  const int tid = threadIdx.x;

  float m = -3.4e38f;
  for (int i = tid; i < N; i += 256) m = fmaxf(m, p[i]);
  red[tid] = m;
  __syncthreads();
  for (int s = 128; s > 0; s >>= 1) {
    if (tid < s) red[tid] = fmaxf(red[tid], red[tid + s]);
    __syncthreads();
  }
  m = red[0];
  __syncthreads();

  float sum = 0.f;
  for (int i = tid; i < N; i += 256) {
    float e = __expf(p[i] - m);
    p[i] = e;
    sum += e;
  }
  red[tid] = sum;
  __syncthreads();
  for (int s = 128; s > 0; s >>= 1) {
    if (tid < s) red[tid] += red[tid + s];
    __syncthreads();
  }
  const float inv = 1.0f / red[0];
  for (int i = tid; i < N; i += 256) o[i] = (bf16_t)(p[i] * inv);
}

// ------------------------------- launcher ----------------------------------
extern "C" void kernel_launch(void* const* d_in, const int* in_sizes, int n_in,
                              void* d_out, int out_size, void* d_ws,
                              size_t ws_size, hipStream_t stream) {
  (void)in_sizes; (void)n_in; (void)out_size; (void)ws_size;

  const float* q  = (const float*)d_in[0];
  const float* wq = (const float*)d_in[1];
  const float* wk = (const float*)d_in[2];
  const float* wv = (const float*)d_in[3];
  const float* wo = (const float*)d_in[4];
  float* out = (float*)d_out;

  const size_t BSD = (size_t)Bn * Sn * Dn;  // 8,388,608
  const size_t DD  = (size_t)Dn * Dn;       // 1,048,576
  const size_t SS  = (size_t)Sn * Sn;       // 4,194,304

  char* ws = (char*)d_ws;
  size_t off = 0;
  bf16_t* qb  = (bf16_t*)(ws + off); off += BSD * 2;          // 16 MiB
  bf16_t* wqb = (bf16_t*)(ws + off); off += DD * 2;           //  2 MiB
  bf16_t* wkb = (bf16_t*)(ws + off); off += DD * 2;
  bf16_t* wvb = (bf16_t*)(ws + off); off += DD * 2;
  bf16_t* wob = (bf16_t*)(ws + off); off += DD * 2;
  bf16_t* Qb  = (bf16_t*)(ws + off); off += BSD * 2;          // 16 MiB
  bf16_t* Kb  = (bf16_t*)(ws + off); off += BSD * 2;          // 16 MiB
  bf16_t* Vt  = (bf16_t*)(ws + off); off += BSD * 2;          // 16 MiB [B][D][S]
  bf16_t* outT= (bf16_t*)(ws + off); off += BSD * 2;          // 16 MiB [B][D][S]
  float*  scb = (float*)(ws + off);  off += SS * 4;           // 16 MiB (per-batch)
  bf16_t* atb = (bf16_t*)(ws + off); off += SS * 2;           //  8 MiB (per-batch)

  // fp32 -> bf16 conversions
  cvt_f32_bf16<<<(int)((BSD + 255) / 256), 256, 0, stream>>>(q, qb, (int)BSD);
  cvt_f32_bf16<<<(int)((DD + 255) / 256), 256, 0, stream>>>(wq, wqb, (int)DD);
  cvt_f32_bf16<<<(int)((DD + 255) / 256), 256, 0, stream>>>(wk, wkb, (int)DD);
  cvt_f32_bf16<<<(int)((DD + 255) / 256), 256, 0, stream>>>(wv, wvb, (int)DD);
  cvt_f32_bf16<<<(int)((DD + 255) / 256), 256, 0, stream>>>(wo, wob, (int)DD);

  // Projections: Q,K row-major [B*S, D]; V transposed per-batch -> Vt[b][e][s]
  dim3 gproj(Dn / 128, (Bn * Sn) / 128);  // (8, 64)
  gemm_bf16_wmma<0><<<gproj, 128, 0, stream>>>(qb, Dn, wqb, Dn, (void*)Qb, Dn,
                                               Dn, 1.0f, 0, 0);
  gemm_bf16_wmma<0><<<gproj, 128, 0, stream>>>(qb, Dn, wkb, Dn, (void*)Kb, Dn,
                                               Dn, 1.0f, 0, 0);
  gemm_bf16_wmma<1><<<gproj, 128, 0, stream>>>(qb, Dn, wvb, Dn, (void*)Vt, Sn,
                                               Dn, 1.0f, Sn,
                                               (long long)Dn * Sn);

  // Per-batch: scores = Q K^T / 8  -> softmax -> outT = (attn V)^T
  for (int b = 0; b < Bn; ++b) {
    const bf16_t* Qbb = Qb + (size_t)b * Sn * Dn;
    const bf16_t* Kbb = Kb + (size_t)b * Sn * Dn;
    const bf16_t* Vtb = Vt + (size_t)b * Dn * Sn;
    bf16_t* oTb = outT + (size_t)b * Dn * Sn;

    dim3 gsc(Sn / 128, Sn / 128);  // (16,16)
    gemm_bf16_wmma<2><<<gsc, 128, 0, stream>>>(Qbb, Dn, Kbb, Dn, (void*)scb,
                                               Sn, Dn, 0.125f, 0, 0);
    softmax_row<<<Sn, 256, 0, stream>>>(scb, atb, Sn);

    dim3 gav(Dn / 128, Sn / 128);  // (8,16)
    gemm_bf16_wmma<1><<<gav, 128, 0, stream>>>(atb, Sn, Vtb, Sn, (void*)oTb,
                                               Sn, Sn, 1.0f, Sn, 0);
  }

  // Final projection: outT flat == reference's buggy reshape of [B,D,S];
  // view as [B*S, D] bf16 A-matrix.  fp32 output.
  gemm_bf16_wmma<2><<<gproj, 128, 0, stream>>>(outT, Dn, wob, Dn, (void*)out,
                                               Dn, Dn, 1.0f, 0, 0);
}
